// CompressedPrototype_73212012527737
// MI455X (gfx1250) — compile-verified, matrix operand order checked by reading, not run on hardware
//
#include <hip/hip_runtime.h>

typedef __attribute__((ext_vector_type(16))) _Float16 v16h;
typedef __attribute__((ext_vector_type(8)))  _Float16 v8h;
typedef __attribute__((ext_vector_type(4)))  _Float16 v4h;
typedef __attribute__((ext_vector_type(8)))  float    v8f;
typedef __attribute__((ext_vector_type(4)))  float    v4f;
typedef __attribute__((ext_vector_type(4)))  int      v4i;

#define M_DIM 4096
#define N_DIM 4096
#define K_DIM 4096
#define BLK_M 128
#define BLK_N 256
#define BLK_K 32
#define LDS_PAD 8
#define LDA (BLK_K + LDS_PAD)        // 40 halves = 80 B per tile row (16B-multiple)
#define NSTEP (K_DIM / BLK_K)

union FragAB {
    v16h v;
    struct { v8h lo, hi; } p;
};

// Per-lane async copy: 16 B global -> LDS, tracked by ASYNCcnt (no VGPR data).
__device__ __forceinline__ void async_copy_b128(unsigned lds_addr, const void* gptr) {
    asm volatile("global_load_async_to_lds_b128 %0, %1, off"
                 :: "v"(lds_addr), "v"(gptr)
                 : "memory");
}
__device__ __forceinline__ void wait_async_all() {
    asm volatile("s_wait_asynccnt 0" ::: "memory");
}
// Generic LDS pointer -> 32-bit LDS byte address (aperture maps addr[31:0] to LDS).
__device__ __forceinline__ unsigned lds_addr_of(const void* p) {
    return (unsigned)(unsigned long long)p;
}

// ---------------------------------------------------------------------------
// Phase 0: Xh = (f16) x   (row-major, same layout)
// ---------------------------------------------------------------------------
__global__ __launch_bounds__(256)
void convert_x_f16(const float* __restrict__ x, _Float16* __restrict__ Xh)
{
    const size_t base = ((size_t)blockIdx.x * 256 + threadIdx.x) * 16;
    #pragma unroll
    for (int i = 0; i < 4; ++i) {
        v4f v = *(const v4f*)(x + base + i * 4);
        v4h h = {(_Float16)v.x, (_Float16)v.y, (_Float16)v.z, (_Float16)v.w};
        *(v4h*)(Xh + base + i * 4) = h;
    }
}

// ---------------------------------------------------------------------------
// Phase 1: Bt[n][k] = (f16) mean[indices[k][n]]  (transposed: B staging and
// fragment loads then share the A-side 16-bit WMMA layout).
// ---------------------------------------------------------------------------
__global__ __launch_bounds__(256)
void build_bt_f16(const int* __restrict__ indices,
                  const float* __restrict__ mean,
                  _Float16* __restrict__ Bt)
{
    __shared__ __align__(16) _Float16 s_mean[K_DIM];         // 8 KB f16 table
    __shared__ __align__(16) _Float16 T[128][128 + LDS_PAD]; // transpose tile

    const int tid = threadIdx.x;
    const int k0  = blockIdx.y * 128;
    const int n0  = blockIdx.x * 128;

    #pragma unroll
    for (int i = 0; i < 4; ++i) {
        const int e = (i * 256 + tid) * 4;
        v4f v = *(const v4f*)(mean + e);
        v4h h = {(_Float16)v.x, (_Float16)v.y, (_Float16)v.z, (_Float16)v.w};
        *(v4h*)(s_mean + e) = h;
    }
    __syncthreads();

    #pragma unroll
    for (int c = 0; c < 16; ++c) {
        const int ch = tid + c * 256;        // 4096 int4 chunks
        const int kk = ch >> 5;
        const int nv = ch & 31;
        v4i id = *(const v4i*)(indices + (size_t)(k0 + kk) * N_DIM + n0 + nv * 4);
        T[nv * 4 + 0][kk] = s_mean[id.x];
        T[nv * 4 + 1][kk] = s_mean[id.y];
        T[nv * 4 + 2][kk] = s_mean[id.z];
        T[nv * 4 + 3][kk] = s_mean[id.w];
    }
    __syncthreads();

    #pragma unroll
    for (int c = 0; c < 8; ++c) {
        const int ch = tid + c * 256;        // 2048 v8h chunks
        const int n  = ch >> 4;
        const int kc = ch & 15;
        v8h v = *(const v8h*)&T[n][kc * 8];
        *(v8h*)(Bt + (size_t)(n0 + n) * K_DIM + k0 + kc * 8) = v;
    }
}

// ---------------------------------------------------------------------------
// Phase 2: C = A x B + bias.  PRE=true: both operands pre-materialized f16,
// staged with GLOBAL_LOAD_ASYNC_TO_LDS_B128 (ASYNCcnt).  PRE=false: fused
// gather fallback.  Block 128x256, 8 waves, 64x64/wave, double-buffered LDS,
// one barrier per K-step.
// ---------------------------------------------------------------------------
template <bool PRE>
__global__ __launch_bounds__(256)
void gather_gemm_wmma(const float* __restrict__ x,
                      const int*   __restrict__ indices,
                      const float* __restrict__ mean,
                      const float* __restrict__ bias,
                      const _Float16* __restrict__ Xh,
                      const _Float16* __restrict__ Bt,
                      float* __restrict__ out)
{
    __shared__ __align__(16) _Float16 s_mean[PRE ? 1 : K_DIM];
    __shared__ __align__(16) _Float16 sA[2][BLK_M][LDA];   // [m][k] f16
    __shared__ __align__(16) _Float16 sB[2][BLK_N][LDA];   // [n][k] f16 (B^T)

    const int tid  = threadIdx.x;
    const int lane = tid & 31;
    const int wave = tid >> 5;
    const int wm   = wave >> 2;
    const int wn   = wave & 3;
    const int half = lane >> 4;
    const int lm   = lane & 15;

    const int blockM = blockIdx.y * BLK_M;
    const int blockN = blockIdx.x * BLK_N;

    if constexpr (!PRE) {
        #pragma unroll
        for (int i = 0; i < 4; ++i) {
            const int e = (i * 256 + tid) * 4;
            v4f v = *(const v4f*)(mean + e);
            v4h h = {(_Float16)v.x, (_Float16)v.y, (_Float16)v.z, (_Float16)v.w};
            *(v4h*)(s_mean + e) = h;
        }
    }

    v8f acc[4][4];
    #pragma unroll
    for (int i = 0; i < 4; ++i)
        #pragma unroll
        for (int j = 0; j < 4; ++j)
            acc[i][j] = v8f{};

    auto stage = [&](int buf, int k0) {
        if constexpr (PRE) {
            // A tile 128x32 halves: 512 b128 chunks, pure async copy.
            #pragma unroll
            for (int c = 0; c < 2; ++c) {
                const int ch  = tid + c * 256;
                const int row = ch >> 2;               // 4 chunks per row
                const int kc  = ch & 3;
                const _Float16* gp = Xh + (size_t)(blockM + row) * K_DIM + k0 + kc * 8;
                __builtin_prefetch(gp + BLK_K, 0, 3);  // tile after next -> GL2
                async_copy_b128(lds_addr_of(&sA[buf][row][kc * 8]), gp);
            }
            // B tile 256x32 halves: 1024 b128 chunks.
            #pragma unroll
            for (int c = 0; c < 4; ++c) {
                const int ch = tid + c * 256;
                const int n  = ch >> 2;
                const int kc = ch & 3;
                const _Float16* gp = Bt + (size_t)(blockN + n) * K_DIM + k0 + kc * 8;
                __builtin_prefetch(gp + BLK_K, 0, 3);
                async_copy_b128(lds_addr_of(&sB[buf][n][kc * 8]), gp);
            }
        } else {
            // A: f32 -> f16 through VGPRs.
            #pragma unroll
            for (int c = 0; c < 4; ++c) {
                const int ch   = tid + c * 256;
                const int row  = ch >> 3;
                const int colv = ch & 7;
                const float* gp = x + (size_t)(blockM + row) * K_DIM + k0 + colv * 4;
                v4f v = *(const v4f*)gp;
                __builtin_prefetch(gp + BLK_K, 0, 3);
                v4h h = {(_Float16)v.x, (_Float16)v.y, (_Float16)v.z, (_Float16)v.w};
                *(v4h*)&sA[buf][row][colv * 4] = h;
            }
            // B: fused gather through the LDS mean table.
            #pragma unroll
            for (int c = 0; c < 8; ++c) {
                const int ch = tid + c * 256;
                const int kk = ch >> 6;
                const int nv = ch & 63;
                const int* ip = indices + (size_t)(k0 + kk) * N_DIM + blockN + nv * 4;
                v4i id = *(const v4i*)ip;
                __builtin_prefetch(ip + (size_t)BLK_K * N_DIM, 0, 3);
                sB[buf][nv * 4 + 0][kk] = s_mean[id.x];
                sB[buf][nv * 4 + 1][kk] = s_mean[id.y];
                sB[buf][nv * 4 + 2][kk] = s_mean[id.z];
                sB[buf][nv * 4 + 3][kk] = s_mean[id.w];
            }
        }
    };

    __syncthreads();          // s_mean ready (fused) / uniform entry
    stage(0, 0);

    int cur = 0;
    for (int kb = 0; kb < NSTEP; ++kb) {
        if constexpr (PRE) wait_async_all();   // my async writes to buf[cur] landed
        __syncthreads();                        // everyone's landed; prior reads done
        if (kb + 1 < NSTEP)
            stage(cur ^ 1, (kb + 1) * BLK_K);

        FragAB a[4], b[4];
        #pragma unroll
        for (int t = 0; t < 4; ++t) {
            const int row = wm * 64 + t * 16 + lm;
            a[t].p.lo = *(const v8h*)&sA[cur][row][half * 8];
            a[t].p.hi = *(const v8h*)&sA[cur][row][16 + half * 8];
        }
        #pragma unroll
        for (int t = 0; t < 4; ++t) {
            const int col = wn * 64 + t * 16 + lm;
            b[t].p.lo = *(const v8h*)&sB[cur][col][half * 8];
            b[t].p.hi = *(const v8h*)&sB[cur][col][16 + half * 8];
        }
        #pragma unroll
        for (int i = 0; i < 4; ++i)
            #pragma unroll
            for (int j = 0; j < 4; ++j)
                acc[i][j] = __builtin_amdgcn_wmma_f32_16x16x32_f16(
                    false, a[i].v, false, b[j].v,
                    (short)0, acc[i][j], false, false);

        cur ^= 1;
    }

    #pragma unroll
    for (int j = 0; j < 4; ++j) {
        const int n  = blockN + wn * 64 + j * 16 + lm;
        const float bv = bias[n];
        #pragma unroll
        for (int i = 0; i < 4; ++i) {
            #pragma unroll
            for (int r = 0; r < 8; ++r) {
                const int m = blockM + wm * 64 + i * 16 + half * 8 + r;
                out[(size_t)m * N_DIM + n] = acc[i][j][r] + bv;
            }
        }
    }
}

extern "C" void kernel_launch(void* const* d_in, const int* in_sizes, int n_in,
                              void* d_out, int out_size, void* d_ws, size_t ws_size,
                              hipStream_t stream) {
    const float* x       = (const float*)d_in[0];
    const int*   indices = (const int*)d_in[1];
    const float* mean    = (const float*)d_in[2];
    const float* bias    = (const float*)d_in[3];
    float*       out     = (float*)d_out;

    const dim3 block(256);
    const dim3 grid(N_DIM / BLK_N, M_DIM / BLK_M);      // 16 x 32

    const size_t bt_bytes = (size_t)K_DIM * N_DIM * sizeof(_Float16);  // 32 MB
    const size_t xh_bytes = (size_t)M_DIM * K_DIM * sizeof(_Float16);  // 32 MB
    if (ws_size >= bt_bytes + xh_bytes) {
        _Float16* Bt = (_Float16*)d_ws;
        _Float16* Xh = (_Float16*)((char*)d_ws + bt_bytes);
        convert_x_f16<<<dim3((M_DIM * K_DIM) / (256 * 16)), block, 0, stream>>>(x, Xh);
        build_bt_f16<<<dim3(N_DIM / 128, K_DIM / 128), block, 0, stream>>>(
            indices, mean, Bt);
        gather_gemm_wmma<true><<<grid, block, 0, stream>>>(
            x, indices, mean, bias, Xh, Bt, out);
    } else {
        gather_gemm_wmma<false><<<grid, block, 0, stream>>>(
            x, indices, mean, bias, nullptr, nullptr, out);
    }
}
